// AttnRNNBlock_58428735094960
// MI455X (gfx1250) — matrix operand order repaired and checked by reading, not compiled, not run on hardware
//
#include <hip/hip_runtime.h>
#include <math.h>

#define HH 1024
#define BB 64
#define SS 128
#define TT 127

typedef __attribute__((ext_vector_type(16))) __bf16 bf16x16;
typedef __attribute__((ext_vector_type(8)))  float  f32x8;

__device__ __forceinline__ unsigned short f2bf(float f) {
  union { float f; unsigned u; } x; x.f = f;
  unsigned r = x.u + 0x7fffu + ((x.u >> 16) & 1u);   // round-to-nearest-even
  return (unsigned short)(r >> 16);
}
__device__ __forceinline__ float bf2f(unsigned short h) {
  union { unsigned u; float f; } x; x.u = ((unsigned)h) << 16;
  return x.f;
}

// ---------------- elementwise helpers ----------------

__global__ void k_cast_bf16(const float* __restrict__ in, unsigned short* __restrict__ out, int n) {
  int i = blockIdx.x * blockDim.x + threadIdx.x;
  if (i < n) out[i] = f2bf(in[i]);
}

__global__ void k_init_h(const float* __restrict__ h0, float* __restrict__ hf,
                         unsigned short* __restrict__ hb) {
  int i = blockIdx.x * blockDim.x + threadIdx.x;   // 65536 threads
  float v = h0[i];
  hf[i] = v;
  hb[i] = f2bf(v);
}

// G[n,k] = sum_j Wih[n,j] * W_w[j,k]   (one-time per layer, f32 VALU, bf16 out)
__global__ void k_fold(const float* __restrict__ Wih, const float* __restrict__ Ww,
                       unsigned short* __restrict__ G) {
  int idx = blockIdx.x * blockDim.x + threadIdx.x;   // HH*HH threads
  int n = idx >> 10, k = idx & (HH - 1);
  float acc = 0.f;
  for (int j = 0; j < HH; ++j) acc = fmaf(Wih[n * HH + j], Ww[j * HH + k], acc);
  G[idx] = f2bf(acc);
}

// bias[n] = bih[n] + bhh[n] + sum_j W_b[j]*Wih[n,j]
__global__ void k_bias(const float* __restrict__ Wih, const float* __restrict__ Wb,
                       const float* __restrict__ bih, const float* __restrict__ bhh,
                       float* __restrict__ biasv) {
  int n = blockIdx.x * blockDim.x + threadIdx.x;
  if (n >= HH) return;
  float acc = bih[n] + bhh[n];
  for (int j = 0; j < HH; ++j) acc = fmaf(Wb[j], Wih[n * HH + j], acc);
  biasv[n] = acc;
}

// ---------------- WMMA fragment loads (ISA 7.12.2 layouts) ----------------

union AF { bf16x16 v; unsigned u[8]; };

// A 16x32 bf16: lanes 0-15 -> rows, K 0..7 in v0-3 / 16..23 in v4-7;
//               lanes 16-31 -> same rows, K 8..15 / 24..31.
__device__ __forceinline__ void load_afrag(AF& a, const unsigned short* __restrict__ base,
                                           int ld, int row, int lane, int k0) {
  int khalf = (lane >= 16) ? 8 : 0;
  const unsigned short* p = base + (size_t)row * ld + k0;
#pragma unroll
  for (int v = 0; v < 8; ++v) {
    int kk = ((v < 4) ? 0 : 16) + khalf + (v & 3) * 2;
    a.u[v] = *(const unsigned*)(p + kk);
  }
}
// B 32x16 bf16 from (N,K)-layout weights: lanes 0-15 col n K=0..7 / 16..23, lanes 16-31 K=8..15 / 24..31.
__device__ __forceinline__ void load_bfrag(AF& b, const unsigned short* __restrict__ base,
                                           int ld, int col, int lane, int k0) {
  int kh = (lane >= 16) ? 8 : 0;
  const unsigned short* p = base + (size_t)col * ld + k0 + kh;
#pragma unroll
  for (int v = 0; v < 8; ++v) b.u[v] = *(const unsigned*)(p + ((v < 4) ? 0 : 16) + (v & 3) * 2);
}

// ---------------- P = X @ Wih^T + bias  (parallel, off critical path) ----------------
// X: (B*T, H) bf16 row-major; Wih: (N,K) bf16; P: (B*T, H) f32
__global__ void __launch_bounds__(128) k_pgemm(const unsigned short* __restrict__ X,
                                               const unsigned short* __restrict__ Wih,
                                               const float* __restrict__ biasv,
                                               float* __restrict__ P) {
  int lane = threadIdx.x & 31;
  int wave = threadIdx.x >> 5;
  int mtile = blockIdx.y * 4 + wave;   // 508 m-tiles
  int ntile = blockIdx.x;              // 64 n-tiles
  int mbase = mtile * 16, nbase = ntile * 16;
  int row = mbase + (lane & 15);
  int col = nbase + (lane & 15);
  f32x8 acc;
  float bv = biasv[col];
#pragma unroll
  for (int r = 0; r < 8; ++r) acc[r] = bv;
  for (int k0 = 0; k0 < HH; k0 += 32) {
    AF a, b;
    load_afrag(a, X, HH, row, lane, k0);
    load_bfrag(b, Wih, HH, col, lane, k0);
    acc = __builtin_amdgcn_wmma_f32_16x16x32_bf16(false, a.v, false, b.v,
                                                  (short)0, acc, false, false);
  }
  int rofs = (lane >= 16) ? 8 : 0;
#pragma unroll
  for (int r = 0; r < 8; ++r) P[(size_t)(mbase + rofs + r) * HH + col] = acc[r];
}

// ---------------- per-step attention: scores + softmax + context ----------------
__global__ void __launch_bounds__(256) k_attn(const float* __restrict__ hf,
                                              const unsigned short* __restrict__ enc,
                                              unsigned short* __restrict__ attn) {
  int b = blockIdx.x;
  int tid = threadIdx.x;
  __shared__ float lh[HH];
  __shared__ float ls[SS];
  __shared__ float red;
  const float* hrow = hf + (size_t)b * HH;
#pragma unroll
  for (int i = 0; i < 4; ++i) lh[tid + 256 * i] = hrow[tid + 256 * i];
  __syncthreads();

  int wave = tid >> 5, lane = tid & 31;
  const unsigned short* encb = enc + (size_t)b * SS * HH;
  for (int j = wave; j < SS; j += 8) {
    const unsigned short* er = encb + (size_t)j * HH;
    float s = 0.f;
    for (int c = lane; c < HH; c += 32) s = fmaf(lh[c], bf2f(er[c]), s);
#pragma unroll
    for (int o = 16; o > 0; o >>= 1) s += __shfl_xor(s, o, 32);
    if (lane == 0) ls[j] = s;
  }
  __syncthreads();

  if (wave == 0) {                       // softmax over 128 logits by one wave
    float m = -3.4e38f;
#pragma unroll
    for (int i = 0; i < 4; ++i) m = fmaxf(m, ls[lane + 32 * i]);
#pragma unroll
    for (int o = 16; o > 0; o >>= 1) m = fmaxf(m, __shfl_xor(m, o, 32));
    float sum = 0.f;
#pragma unroll
    for (int i = 0; i < 4; ++i) {
      float e = __expf(ls[lane + 32 * i] - m);
      ls[lane + 32 * i] = e;
      sum += e;
    }
#pragma unroll
    for (int o = 16; o > 0; o >>= 1) sum += __shfl_xor(sum, o, 32);
    if (lane == 0) red = 1.f / sum;
  }
  __syncthreads();
  float inv = red;
#pragma unroll
  for (int i = 0; i < 4; ++i) {
    int c = tid + 256 * i;
    float acc = 0.f;
    for (int j = 0; j < SS; ++j) acc = fmaf(ls[j], bf2f(encb[(size_t)j * HH + c]), acc);
    attn[(size_t)b * HH + c] = f2bf(acc * inv);
  }
}

// ---------------- per-step fused cell: h' = tanh(P_t + attn@G^T + h@Whh^T) ----------------
__global__ void __launch_bounds__(128) k_step(const float* __restrict__ P, int t,
                                              const unsigned short* __restrict__ attn,
                                              const unsigned short* __restrict__ hb_in,
                                              const unsigned short* __restrict__ G,
                                              const unsigned short* __restrict__ Whh,
                                              float* __restrict__ hf_out,
                                              unsigned short* __restrict__ hb_out,
                                              unsigned short* __restrict__ outbf,
                                              float* __restrict__ outf) {
  int lane = threadIdx.x & 31;
  int wave = threadIdx.x >> 5;
  int lin = blockIdx.x * 4 + wave;      // 256 tiles
  int mtile = lin & 3, ntile = lin >> 2;
  int mbase = mtile * 16, nbase = ntile * 16;
  int row = mbase + (lane & 15);
  int col = nbase + (lane & 15);
  int rofs = (lane >= 16) ? 8 : 0;
  f32x8 acc;
#pragma unroll
  for (int r = 0; r < 8; ++r)
    acc[r] = P[((size_t)(mbase + rofs + r) * TT + t) * HH + col];

  for (int k0 = 0; k0 < HH; k0 += 32) {
    AF a, b;
    load_afrag(a, attn, HH, row, lane, k0);
    load_bfrag(b, G, HH, col, lane, k0);
    acc = __builtin_amdgcn_wmma_f32_16x16x32_bf16(false, a.v, false, b.v,
                                                  (short)0, acc, false, false);
  }
  for (int k0 = 0; k0 < HH; k0 += 32) {
    AF a, b;
    load_afrag(a, hb_in, HH, row, lane, k0);
    load_bfrag(b, Whh, HH, col, lane, k0);
    acc = __builtin_amdgcn_wmma_f32_16x16x32_bf16(false, a.v, false, b.v,
                                                  (short)0, acc, false, false);
  }
#pragma unroll
  for (int r = 0; r < 8; ++r) {
    int m = mbase + rofs + r;
    float v = tanhf(acc[r]);
    hf_out[(size_t)m * HH + col] = v;
    unsigned short bv = f2bf(v);
    hb_out[(size_t)m * HH + col] = bv;
    if (outbf) outbf[((size_t)m * TT + t) * HH + col] = bv;
    if (outf)  outf [((size_t)m * TT + t) * HH + col] = v;
  }
}

// ---------------- host orchestration ----------------

static inline size_t alignup(size_t x) { return (x + 255) & ~(size_t)255; }

extern "C" void kernel_launch(void* const* d_in, const int* in_sizes, int n_in,
                              void* d_out, int out_size, void* d_ws, size_t ws_size,
                              hipStream_t stream) {
  const float* enc_out  = (const float*)d_in[0];   // (B,S,H)
  const float* x0       = (const float*)d_in[1];   // (B,T,H)
  const float* h0_all   = (const float*)d_in[2];   // (L,B,H)
  const float* W_w      = (const float*)d_in[3];   // (H,H)
  const float* W_b      = (const float*)d_in[4];   // (H,)
  const float* Wih_all  = (const float*)d_in[5];   // (L,H,H)
  const float* Whh_all  = (const float*)d_in[6];   // (L,H,H)
  const float* bih_all  = (const float*)d_in[7];   // (L,H)
  const float* bhh_all  = (const float*)d_in[8];   // (L,H)
  float* out = (float*)d_out;                      // (B,T,H)

  char* ws = (char*)d_ws;
  size_t off = 0;
  auto take = [&](size_t bytes) { char* p = ws + off; off += alignup(bytes); return p; };

  unsigned short* enc_bf = (unsigned short*)take((size_t)BB * SS * HH * 2);
  unsigned short* X0_bf  = (unsigned short*)take((size_t)BB * TT * HH * 2);
  unsigned short* O0_bf  = (unsigned short*)take((size_t)BB * TT * HH * 2);
  float*          P      = (float*)         take((size_t)BB * TT * HH * 4);
  unsigned short* G_bf   = (unsigned short*)take((size_t)HH * HH * 2);
  unsigned short* Whh_bf = (unsigned short*)take((size_t)HH * HH * 2);
  unsigned short* Wih_bf = (unsigned short*)take((size_t)HH * HH * 2);
  float*          biasv  = (float*)         take((size_t)HH * 4);
  unsigned short* attn   = (unsigned short*)take((size_t)BB * HH * 2);
  float*          hf0    = (float*)         take((size_t)BB * HH * 4);
  float*          hf1    = (float*)         take((size_t)BB * HH * 4);
  unsigned short* hb0    = (unsigned short*)take((size_t)BB * HH * 2);
  unsigned short* hb1    = (unsigned short*)take((size_t)BB * HH * 2);
  (void)ws_size; (void)n_in; (void)in_sizes; (void)out_size;
  float* hf[2] = {hf0, hf1};
  unsigned short* hb[2] = {hb0, hb1};

  // one-time: compress encoder states to bf16 (resident in L2 thereafter)
  k_cast_bf16<<<(BB * SS * HH) / 256, 256, 0, stream>>>(enc_out, enc_bf, BB * SS * HH);

  for (int l = 0; l < 2; ++l) {
    const float* Wih = Wih_all + (size_t)l * HH * HH;
    const float* Whh = Whh_all + (size_t)l * HH * HH;

    // per-layer folds (parallel, off critical path)
    k_cast_bf16<<<(HH * HH) / 256, 256, 0, stream>>>(Wih, Wih_bf, HH * HH);
    k_cast_bf16<<<(HH * HH) / 256, 256, 0, stream>>>(Whh, Whh_bf, HH * HH);
    k_fold<<<(HH * HH) / 256, 256, 0, stream>>>(Wih, W_w, G_bf);
    k_bias<<<4, 256, 0, stream>>>(Wih, W_b, bih_all + l * HH, bhh_all + l * HH, biasv);

    const unsigned short* Xbf;
    if (l == 0) {
      k_cast_bf16<<<(BB * TT * HH) / 256, 256, 0, stream>>>(x0, X0_bf, BB * TT * HH);
      Xbf = X0_bf;
    } else {
      Xbf = O0_bf;   // layer-0 output already written in bf16
    }
    // P = X @ Wih^T + (bih + bhh + W_b @ Wih^T)   -- one big WMMA GEMM
    k_pgemm<<<dim3(64, 127), 128, 0, stream>>>(Xbf, Wih_bf, biasv, P);

    k_init_h<<<(BB * HH) / 256, 256, 0, stream>>>(h0_all + (size_t)l * BB * HH, hf[0], hb[0]);

    // sequential recurrence
    for (int t = 0; t < TT; ++t) {
      int pi = t & 1, po = (t + 1) & 1;
      k_attn<<<BB, 256, 0, stream>>>(hf[pi], enc_bf, attn);
      k_step<<<64, 128, 0, stream>>>(P, t, attn, hb[pi], G_bf, Whh_bf,
                                     hf[po], hb[po],
                                     (l == 0) ? O0_bf : nullptr,
                                     (l == 1) ? out : nullptr);
    }
  }
}
